// PFSPNetDecoder_75754633166939
// MI455X (gfx1250) — matrix operand order repaired and checked by reading, not compiled, not run on hardware
//
#include <hip/hip_runtime.h>
#include <hip/hip_bf16.h>

typedef __attribute__((ext_vector_type(8)))  float  v8f;
typedef __attribute__((ext_vector_type(16))) __bf16 v16bf;

#define TB 8             // batch rows per workgroup (padded to 16-row WMMA tiles)
#define NWG 64
#define NB 512
#define NN 200
#define NM 20

// ---------------- math helpers (trans-op based) ----------------
__device__ inline float fast_exp(float x) { return __builtin_amdgcn_exp2f(x * 1.4426950408889634f); }
__device__ inline float fast_log(float x) { return __builtin_amdgcn_logf(x) * 0.6931471805599453f; }
__device__ inline float tanh_fast(float x) {
#if __has_builtin(__builtin_amdgcn_tanhf)
  return __builtin_amdgcn_tanhf(x);            // native v_tanh_f32 on gfx1250
#elif __has_builtin(__builtin_amdgcn_tanh_f32)
  return __builtin_amdgcn_tanh_f32(x);
#else
  float e = __builtin_amdgcn_exp2f(x * 2.8853900817779268f);
  return 1.0f - 2.0f * __builtin_amdgcn_rcpf(e + 1.0f);
#endif
}
__device__ inline unsigned hash3(unsigned a, unsigned b, unsigned c) {
  unsigned h = a * 0x9E3779B9u ^ b * 0x85EBCA6Bu ^ c * 0xC2B2AE35u;
  h ^= h >> 16; h *= 0x7FEB352Du; h ^= h >> 15; h *= 0x846CA68Bu; h ^= h >> 16;
  return h;
}
__device__ inline float bfLo(unsigned u) { return __uint_as_float(u << 16); }
__device__ inline float bfHi(unsigned u) { return __uint_as_float(u & 0xFFFF0000u); }

// ---------------- WMMA helpers ----------------
__device__ inline v8f wmma_bf16(v16bf a, v16bf b, v8f c) {
  return __builtin_amdgcn_wmma_f32_16x16x32_bf16(false, a, false, b, (short)0, c, false, false);
}

// A fragment (16x32 bf16) from f32 row-major [16 x ld] (LDS or global), b128 vector loads.
// lanes0-15: row=lane, K={0..7,16..23}; lanes16-31: row=lane-16, K={8..15,24..31}.
__device__ inline v16bf afrag_pack(const float* xr) {
  float4 a0 = *(const float4*)(xr);
  float4 a1 = *(const float4*)(xr + 4);
  float4 a2 = *(const float4*)(xr + 16);
  float4 a3 = *(const float4*)(xr + 20);
  v16bf a;
  a[0]=(__bf16)a0.x; a[1]=(__bf16)a0.y; a[2]=(__bf16)a0.z; a[3]=(__bf16)a0.w;
  a[4]=(__bf16)a1.x; a[5]=(__bf16)a1.y; a[6]=(__bf16)a1.z; a[7]=(__bf16)a1.w;
  a[8]=(__bf16)a2.x; a[9]=(__bf16)a2.y; a[10]=(__bf16)a2.z; a[11]=(__bf16)a2.w;
  a[12]=(__bf16)a3.x; a[13]=(__bf16)a3.y; a[14]=(__bf16)a3.z; a[15]=(__bf16)a3.w;
  return a;
}
__device__ inline v16bf afrag_from(const float* X, int ld, int ktile, int lane) {
  const int row = lane & 15;
  const int kb  = (lane & 16) ? 8 : 0;
  return afrag_pack(X + row * ld + ktile * 32 + kb);
}
// concat([16x128] H, [16x32] E), K=160: ktile 4 is the E part
__device__ inline v16bf afrag_cat(const float* H, const float* E, int ktile, int lane) {
  const int row = lane & 15;
  const int kb  = (lane & 16) ? 8 : 0;
  if (ktile < 4) return afrag_pack(H + row * 128 + ktile * 32 + kb);
  return afrag_pack(E + row * 32 + kb);
}
// Packed B tile: 512 bf16 per tile, lane-major -> straight 32B vector load
__device__ inline v16bf bfrag(const __bf16* base, int tile, int lane) {
  return *(const v16bf*)(base + (size_t)tile * 512 + lane * 16);
}

// ---------------- weight packer: W[Nout,Kin] f32 -> bf16 B-fragment tiles ----------------
__global__ __launch_bounds__(256) void pack_weight_kernel(const float* __restrict__ W,
                                                          __bf16* __restrict__ dst,
                                                          int Nout, int Kin) {
  const int NT = Nout >> 4;
  const int total = (Kin / 32) * NT * 512;
  for (int idx = blockIdx.x * 256 + threadIdx.x; idx < total; idx += gridDim.x * 256) {
    int tile = idx >> 9;
    int lane = (idx >> 4) & 31;
    int i    = idx & 15;
    int kt = tile / NT, nt = tile - kt * NT;
    int n = nt * 16 + (lane & 15);
    int k = kt * 32 + ((lane & 16) ? 8 : 0) + (i < 8 ? i : i + 8);
    dst[idx] = (__bf16)W[n * Kin + k];
  }
}

// ---------------- pre_att = bf16(encoded_jobs @ att_W1^T + b1) ----------------
__global__ __launch_bounds__(256) void preatt_kernel(const float* __restrict__ ej,
                                                     const __bf16* __restrict__ wb,
                                                     const float* __restrict__ b1,
                                                     __bf16* __restrict__ pre) {
  const int lane = threadIdx.x & 31;
  const int wid  = (blockIdx.x * 256 + threadIdx.x) >> 5;  // 102400 waves
  const int mt = wid >> 4;                                  // 6400 row tiles
  const int nt = wid & 15;                                  // 16 col tiles
  v8f acc = {};
#pragma unroll
  for (int kt = 0; kt < 8; ++kt) {
    v16bf a = afrag_from(ej + (size_t)mt * 16 * 256, 256, kt, lane);
    v16bf b = bfrag(wb, kt * 16 + nt, lane);
    acc = wmma_bf16(a, b, acc);
  }
  const int col = nt * 16 + (lane & 15);
  const float bias = b1[col];
  const int rb = (lane & 16) ? 8 : 0;
#pragma unroll
  for (int v = 0; v < 8; ++v) {
    int grow = mt * 16 + rb + v;
    pre[(size_t)grow * 256 + col] = (__bf16)(acc[v] + bias);
  }
}

// ---------------- persistent decoder: 8 batch rows / WG (16-row padded tiles), 200 steps ----------------
__global__ __launch_bounds__(256) void decode_kernel(
    const float* __restrict__ pt,     // [B,N,M,2]
    const float* __restrict__ nm,     // [B]
    const float* __restrict__ wih1,   // [128,2]
    const float* __restrict__ b1,     // [128]
    const float* __restrict__ mw, const float* __restrict__ mb,  // [32]
    const float* __restrict__ fcb,    // [128]
    const float* __restrict__ d0,     // [256]
    const float* __restrict__ b2,     // [256]
    const float* __restrict__ attb2,  // [256]
    const float* __restrict__ attv,   // [256]
    const __bf16* __restrict__ wpack,
    const __bf16* __restrict__ preatt,
    float* __restrict__ out)
{
  // 16-row padded state (rows 8..15 are deterministic padding, never stored out)
  __shared__ float    sh_h2[16 * 256];
  __shared__ float    sh_P[16 * 40];
  __shared__ float    sh_memb[16 * 32];
  __shared__ unsigned sh_avail[TB * 8];
  __shared__ float    sh_v[256];
  __shared__ int      sh_chosen[TB];
  __shared__ float    arena[10240];   // 40KB overlay

  float* di  = arena;           // [16x128]
  float* h1a = arena + 2048;    // [16x128]
  float* h1b = arena + 4096;    // [16x128]
  float* h2n = arena + 2048;    // [16x256] (after rnn1/fc dead)
  float* q   = arena + 6144;    // [16x256]
  float* scr = arena;           // [8x200]  (after di/h2n dead)

  const int tid  = threadIdx.x;
  const int lane = tid & 31;
  const int wave = tid >> 5;
  const int b0   = blockIdx.x * TB;

  const __bf16* whh1  = wpack + (size_t)0   * 512;
  const __bf16* fcw   = wpack + (size_t)32  * 512;
  const __bf16* wih2  = wpack + (size_t)72  * 512;
  const __bf16* whh2  = wpack + (size_t)136 * 512;
  const __bf16* attw2 = wpack + (size_t)264 * 512;

  for (int i = tid; i < 16 * 256; i += 256) sh_h2[i] = d0[i & 255];
  for (int i = tid; i < 16 * 40;  i += 256) sh_P[i] = 0.f;
  for (int i = tid; i < 16 * 32;  i += 256) {
    int r = i >> 5, e = i & 31;
    sh_memb[i] = (r < TB) ? (nm[b0 + r] * mw[e] + mb[e]) : 0.f;
  }
  for (int i = tid; i < TB * 8;   i += 256) { int w = i & 7; sh_avail[i] = (w < 6) ? 0xFFFFFFFFu : (w == 6 ? 0xFFu : 0u); }
  if (tid < 256) sh_v[tid] = attv[tid];
  __syncthreads();

  for (int t = 0; t < NN; ++t) {
    // ---- step1: RNN over 20 machines, h:[16,128] via WMMA, ntile = wave ----
    for (int i = tid; i < 16 * 128; i += 256) h1a[i] = 0.f;
    __syncthreads();
    float* cur = h1a; float* nxt = h1b;
    for (int m = 0; m < NM; ++m) {
      v8f acc = {};
#pragma unroll
      for (int kt = 0; kt < 4; ++kt)
        acc = wmma_bf16(afrag_from(cur, 128, kt, lane), bfrag(whh1, kt * 8 + wave, lane), acc);
      const int col = wave * 16 + (lane & 15);
      const float wx0 = wih1[col * 2 + 0], wx1 = wih1[col * 2 + 1];
      const float bias = b1[col];
      const int rb = (lane & 16) ? 8 : 0;
#pragma unroll
      for (int v = 0; v < 8; ++v) {
        int r = rb + v;
        float x0 = sh_P[r * 40 + m * 2 + 0], x1 = sh_P[r * 40 + m * 2 + 1];
        nxt[r * 128 + col] = tanh_fast(acc[v] + bias + x0 * wx0 + x1 * wx1);
      }
      __syncthreads();
      float* tmp = cur; cur = nxt; nxt = tmp;
    }
    // ---- fc: di = concat(h, m_emb)[16,160] @ fc_W^T + fcb ----
    {
      v8f acc = {};
#pragma unroll
      for (int kt = 0; kt < 5; ++kt)
        acc = wmma_bf16(afrag_cat(cur, sh_memb, kt, lane), bfrag(fcw, kt * 8 + wave, lane), acc);
      const int col = wave * 16 + (lane & 15);
      const float bias = fcb[col];
      const int rb = (lane & 16) ? 8 : 0;
#pragma unroll
      for (int v = 0; v < 8; ++v) di[(rb + v) * 128 + col] = acc[v] + bias;
    }
    __syncthreads();
    // ---- rnn2: h2n = tanh(di@Wih2^T + h2@Whh2^T + b2), 16 N-tiles => 2/wave ----
    for (int half = 0; half < 2; ++half) {
      const int nt = wave + half * 8;
      v8f acc = {};
#pragma unroll
      for (int kt = 0; kt < 4; ++kt)
        acc = wmma_bf16(afrag_from(di, 128, kt, lane), bfrag(wih2, kt * 16 + nt, lane), acc);
#pragma unroll
      for (int kt = 0; kt < 8; ++kt)
        acc = wmma_bf16(afrag_from(sh_h2, 256, kt, lane), bfrag(whh2, kt * 16 + nt, lane), acc);
      const int col = nt * 16 + (lane & 15);
      const float bias = b2[col];
      const int rb = (lane & 16) ? 8 : 0;
#pragma unroll
      for (int v = 0; v < 8; ++v) h2n[(rb + v) * 256 + col] = tanh_fast(acc[v] + bias);
    }
    __syncthreads();
    // ---- q = h2n @ attW2^T + attb2 ----
    for (int half = 0; half < 2; ++half) {
      const int nt = wave + half * 8;
      v8f acc = {};
#pragma unroll
      for (int kt = 0; kt < 8; ++kt)
        acc = wmma_bf16(afrag_from(h2n, 256, kt, lane), bfrag(attw2, kt * 16 + nt, lane), acc);
      const int col = nt * 16 + (lane & 15);
      const float bias = attb2[col];
      const int rb = (lane & 16) ? 8 : 0;
#pragma unroll
      for (int v = 0; v < 8; ++v) q[(rb + v) * 256 + col] = acc[v] + bias;
    }
    __syncthreads();
    for (int i = tid; i < 16 * 256; i += 256) sh_h2[i] = h2n[i];   // h2 <- h2n (frees h2n region)
    __syncthreads();
    // ---- attention scores, fully vectorized b128 loads (pre_att bf16 from L2) ----
    for (int p = tid; p < TB * NN; p += 256) {
      const int r = p / NN, n = p - r * NN;
      const __bf16* pa = preatt + ((size_t)(b0 + r) * NN + n) * 256;
      const float* qr = q + r * 256;
      float s = 0.f;
#pragma unroll 2
      for (int kk = 0; kk < 256; kk += 8) {
        uint4  u  = *(const uint4*)(pa + kk);
        float4 q0 = *(const float4*)(qr + kk);
        float4 q1 = *(const float4*)(qr + kk + 4);
        float4 v0 = *(const float4*)(sh_v + kk);
        float4 v1 = *(const float4*)(sh_v + kk + 4);
        s += tanh_fast(bfLo(u.x) + q0.x) * v0.x + tanh_fast(bfHi(u.x) + q0.y) * v0.y;
        s += tanh_fast(bfLo(u.y) + q0.z) * v0.z + tanh_fast(bfHi(u.y) + q0.w) * v0.w;
        s += tanh_fast(bfLo(u.z) + q1.x) * v1.x + tanh_fast(bfHi(u.z) + q1.y) * v1.y;
        s += tanh_fast(bfLo(u.w) + q1.z) * v1.z + tanh_fast(bfHi(u.w) + q1.w) * v1.w;
      }
      scr[r * NN + n] = s;
    }
    __syncthreads();
    // ---- masked softmax + logp + gumbel sample: one wave per row ----
    {
      const int r  = wave;               // 8 waves, 8 rows
      const int bg = b0 + r;
      float vals[7];
      float mx = -3e38f;
#pragma unroll
      for (int j = 0; j < 7; ++j) {
        int n = lane + j * 32;
        float mv = -1e9f;
        if (n < NN) {
          bool av = (sh_avail[r * 8 + (n >> 5)] >> (n & 31)) & 1u;
          mv = av ? scr[r * NN + n] : -1e9f;
        }
        vals[j] = mv;
        mx = fmaxf(mx, mv);
      }
#pragma unroll
      for (int mk = 1; mk < 32; mk <<= 1) mx = fmaxf(mx, __shfl_xor(mx, mk, 32));
      float sum = 0.f;
#pragma unroll
      for (int j = 0; j < 7; ++j) { int n = lane + j * 32; if (n < NN) sum += fast_exp(vals[j] - mx); }
#pragma unroll
      for (int mk = 1; mk < 32; mk <<= 1) sum += __shfl_xor(sum, mk, 32);
      const float lse = mx + fast_log(sum);
      float best = -3e38f; int bestn = 0;
      float* lp = out + 102400 + ((size_t)bg * NN + t) * NN;
#pragma unroll
      for (int j = 0; j < 7; ++j) {
        int n = lane + j * 32;
        if (n < NN) {
          lp[n] = fast_log(fast_exp(vals[j] - lse) + 1e-9f);
          unsigned h = hash3(42u ^ (unsigned)t, (unsigned)bg, (unsigned)n);
          float u = ((float)h + 0.5f) * (1.0f / 4294967296.0f);
          float g = -fast_log(-fast_log(u));
          float cand = vals[j] + g;
          if (cand > best) { best = cand; bestn = n; }
        }
      }
#pragma unroll
      for (int mk = 1; mk < 32; mk <<= 1) {
        float ob = __shfl_xor(best, mk, 32);
        int   on = __shfl_xor(bestn, mk, 32);
        if (ob > best || (ob == best && on < bestn)) { best = ob; bestn = on; }
      }
      if (lane == 0) { sh_chosen[r] = bestn; out[(size_t)bg * NN + t] = (float)bestn; }
    }
    __syncthreads();
    // ---- state update: clear avail bit, gather P_next (rows 0..7 only) ----
    if (tid < TB) {
      int c = sh_chosen[tid];
      sh_avail[tid * 8 + (c >> 5)] &= ~(1u << (c & 31));
    }
    for (int i = tid; i < TB * 40; i += 256) {
      int r = i / 40, e = i - r * 40;
      int c = sh_chosen[r];
      sh_P[i] = pt[((size_t)(b0 + r) * NN + c) * 40 + e];
    }
    __syncthreads();
  }
}

// ---------------- host launch ----------------
extern "C" void kernel_launch(void* const* d_in, const int* in_sizes, int n_in,
                              void* d_out, int out_size, void* d_ws, size_t ws_size,
                              hipStream_t stream) {
  const float* ej   = (const float*)d_in[0];   // [512,200,256]
  const float* pt   = (const float*)d_in[1];   // [512,200,20,2]
  const float* nm   = (const float*)d_in[2];   // [512]
  const float* wih1 = (const float*)d_in[3];   // [128,2]
  const float* whh1 = (const float*)d_in[4];   // [128,128]
  const float* rb1  = (const float*)d_in[5];   // [128]
  const float* mw   = (const float*)d_in[6];   // [32]
  const float* mbv  = (const float*)d_in[7];   // [32]
  const float* fcW  = (const float*)d_in[8];   // [128,160]
  const float* fcb  = (const float*)d_in[9];   // [128]
  const float* d0   = (const float*)d_in[10];  // [256]
  const float* wih2 = (const float*)d_in[11];  // [256,128]
  const float* whh2 = (const float*)d_in[12];  // [256,256]
  const float* rb2  = (const float*)d_in[13];  // [256]
  const float* aW1  = (const float*)d_in[14];  // [256,256]
  const float* ab1  = (const float*)d_in[15];  // [256]
  const float* aW2  = (const float*)d_in[16];  // [256,256]
  const float* ab2  = (const float*)d_in[17];  // [256]
  const float* av   = (const float*)d_in[18];  // [256]

  __bf16* wpack  = (__bf16*)d_ws;
  __bf16* preatt = (__bf16*)((char*)d_ws + (1u << 20));
  float*  out    = (float*)d_out;

  // pack weights into WMMA B-fragment tiles (tile offsets in units of 512 bf16)
  pack_weight_kernel<<<64,  256, 0, stream>>>(whh1, wpack + (size_t)0   * 512, 128, 128); // 32 tiles
  pack_weight_kernel<<<80,  256, 0, stream>>>(fcW,  wpack + (size_t)32  * 512, 128, 160); // 40 tiles
  pack_weight_kernel<<<128, 256, 0, stream>>>(wih2, wpack + (size_t)72  * 512, 256, 128); // 64 tiles
  pack_weight_kernel<<<256, 256, 0, stream>>>(whh2, wpack + (size_t)136 * 512, 256, 256); // 128 tiles
  pack_weight_kernel<<<256, 256, 0, stream>>>(aW2,  wpack + (size_t)264 * 512, 256, 256); // 128 tiles
  pack_weight_kernel<<<256, 256, 0, stream>>>(aW1,  wpack + (size_t)392 * 512, 256, 256); // 128 tiles

  // pre_att (bf16, L2-resident across the 200 decode steps): 6400x16 tiles, 8 waves/WG
  preatt_kernel<<<12800, 256, 0, stream>>>(ej, wpack + (size_t)392 * 512, ab1, preatt);

  // persistent decoder: 64 WGs x 8 batch rows
  decode_kernel<<<NWG, 256, 0, stream>>>(pt, nm, wih1, rb1, mw, mbv, fcb, d0, rb2,
                                         ab2, av, wpack, preatt, out);

  // third output: encoded_jobs passthrough
  hipMemcpyAsync(out + 102400 + 20480000, ej, (size_t)26214400 * sizeof(float),
                 hipMemcpyDeviceToDevice, stream);
}
